// DotProductAttention_22041772163235
// MI455X (gfx1250) — matrix operand order repaired and checked by reading, not compiled, not run on hardware
//
#include <hip/hip_runtime.h>

// DotProductAttention, B=2 H=16 C=128 Lq=Lkv=2048, fp32, softmax over the
// *query* axis (per key-column normalization).
//
// Pipeline (all on `stream`, serialized by stream order):
//  1) transpose_split : Q,K -> transposed bf16 hi/lo (Qt[q][c], Kt[k][c])
//  2) qk_colstats     : per-column m[k] = max_q S, 1/d[k] = 1/sum_q exp(S-m)
//  3) v_scale_split   : V*(1/d[k]) -> bf16 hi/lo (contraction axis k contiguous)
//  4) attn_out        : O[c,q] = sum_k V'[c,k] * exp(S[q,k]-m[k]), S recomputed
// All GEMMs: v_wmma_f32_16x16x32_bf16 with hi/lo splitting (3 WMMAs/tile,
// ~fp32 accuracy at bf16-pipe rate). Splitting is hoisted out of hot loops.

#define LEN   2048
#define CH    128
#define NHEAD 32   // B*H

typedef __attribute__((ext_vector_type(16))) __bf16 v16bf;
typedef __attribute__((ext_vector_type(8)))  __bf16 bf16x8;
typedef __attribute__((ext_vector_type(8)))  float  v8f;

#define ZERO8 {0.f,0.f,0.f,0.f,0.f,0.f,0.f,0.f}

__device__ __forceinline__ v8f wmma_bf16(v16bf a, v16bf b, v8f c) {
  return __builtin_amdgcn_wmma_f32_16x16x32_bf16(
      false, a, false, b, (short)0, c, false, false);
}

// (ah+al)*(bh+bl) ~= ah*bh + ah*bl + al*bh
__device__ __forceinline__ v8f wmma_split(v16bf ah, v16bf al,
                                          v16bf bh, v16bf bl, v8f acc) {
  acc = wmma_bf16(al, bh, acc);
  acc = wmma_bf16(ah, bl, acc);
  acc = wmma_bf16(ah, bh, acc);
  return acc;
}

// A-fragment element map (16-bit A 16x32): with p = row_base + lhi*8,
// elements 0..7 = p[0..7], elements 8..15 = p[16..23]. Two b128 loads.
__device__ __forceinline__ v16bf loadA16(const __bf16* p) {
  bf16x8 a = *(const bf16x8*)p;
  bf16x8 b = *(const bf16x8*)(p + 16);
  return __builtin_shufflevector(a, b, 0, 1, 2, 3, 4, 5, 6, 7,
                                 8, 9, 10, 11, 12, 13, 14, 15);
}

// ---------------------------------------------------------------------------
// 1) Transpose + bf16 hi/lo split of Q and K: [c][l] f32 -> [l][c] bf16 x2.
//    grid (LEN/64, NHEAD, 2), block 256. LDS tile, coalesced on both sides.
// ---------------------------------------------------------------------------
__global__ __launch_bounds__(256) void transpose_split(
    const float* __restrict__ Qsrc, const float* __restrict__ Ksrc,
    __bf16* __restrict__ Qth, __bf16* __restrict__ Qtl,
    __bf16* __restrict__ Kth, __bf16* __restrict__ Ktl) {
  __shared__ float tile[CH][65];  // pad -> conflict-free column reads
  const int head = blockIdx.y;
  const int l0 = blockIdx.x * 64;
  const size_t hoff = (size_t)head * CH * LEN;
  const float* src = blockIdx.z ? Ksrc : Qsrc;
  __bf16* dh = blockIdx.z ? Kth : Qth;
  __bf16* dl = blockIdx.z ? Ktl : Qtl;
  const int tid = threadIdx.x;
  {
    const int ll = tid & 63, cb = tid >> 6;
#pragma unroll
    for (int i = 0; i < 32; ++i) {
      int c = i * 4 + cb;
      tile[c][ll] = src[hoff + (size_t)c * LEN + l0 + ll];
    }
  }
  __syncthreads();
  {
    const int c = tid & 127, lb = tid >> 7;
#pragma unroll
    for (int i = 0; i < 32; ++i) {
      int ll = i * 2 + lb;
      float x = tile[c][ll];
      __bf16 h = (__bf16)x;
      size_t o = hoff + (size_t)(l0 + ll) * CH + c;
      dh[o] = h;
      dl[o] = (__bf16)(x - (float)h);
    }
  }
}

// ---------------------------------------------------------------------------
// 2) Column stats of S = Q^T K. grid (LEN/128, NHEAD), block 256 (8 waves).
//    Each wave: 16 k-columns, K B-fragments hoisted, streams all q online.
// ---------------------------------------------------------------------------
__global__ __launch_bounds__(256) void qk_colstats(
    const __bf16* __restrict__ Qth, const __bf16* __restrict__ Qtl,
    const __bf16* __restrict__ Kth, const __bf16* __restrict__ Ktl,
    float* __restrict__ colmax, float* __restrict__ colrcp) {
  const int wave = threadIdx.x >> 5;
  const int lane = threadIdx.x & 31;
  const int lrow = lane & 15;
  const int lhi  = lane >> 4;
  const int head = blockIdx.y;
  const int k0   = blockIdx.x * 128 + wave * 16;
  const size_t hoff = (size_t)head * CH * LEN;

  // B fragments (32x16): lane -> col n=lrow, element e -> c = cc*32+lhi*8+(e&7)+(e>>3)*16
  v16bf bh[4], bl[4];
  const size_t krow = hoff + (size_t)(k0 + lrow) * CH;
#pragma unroll
  for (int cc = 0; cc < 4; ++cc) {
    bh[cc] = loadA16(&Kth[krow + cc * 32 + lhi * 8]);
    bl[cc] = loadA16(&Ktl[krow + cc * 32 + lhi * 8]);
  }

  float m = -3.0e38f, d = 0.f;
  for (int q0 = 0; q0 < LEN; q0 += 16) {
    const size_t qrow = hoff + (size_t)(q0 + lrow) * CH + lhi * 8;
    v8f acc = ZERO8;
#pragma unroll
    for (int cc = 0; cc < 4; ++cc) {
      v16bf ah = loadA16(&Qth[qrow + cc * 32]);
      v16bf al = loadA16(&Qtl[qrow + cc * 32]);
      acc = wmma_split(ah, al, bh[cc], bl[cc], acc);
    }
    // acc[r] = S[q0 + lhi*8 + r, k0 + lrow]
#pragma unroll
    for (int r = 0; r < 8; ++r) {
      float s  = acc[r];
      float nm = fmaxf(m, s);
      d = d * __expf(m - nm) + __expf(s - nm);
      m = nm;
    }
  }
  // lanes L and L+16 hold the same column
  float om = __shfl_xor(m, 16, 32);
  float od = __shfl_xor(d, 16, 32);
  float nm = fmaxf(m, om);
  float dd = d * __expf(m - nm) + od * __expf(om - nm);
  if (lhi == 0) {
    int idx = head * LEN + k0 + lrow;
    colmax[idx] = nm;
    colrcp[idx] = 1.0f / dd;
  }
}

// ---------------------------------------------------------------------------
// 3) V' = V * (1/d[k]) -> bf16 hi/lo. One thread per element, coalesced.
// ---------------------------------------------------------------------------
__global__ __launch_bounds__(256) void v_scale_split(
    const float* __restrict__ V, const float* __restrict__ colrcp,
    __bf16* __restrict__ Vh, __bf16* __restrict__ Vl) {
  int i = blockIdx.x * 256 + threadIdx.x;  // < 2^23, fits int
  int kcol = i & (LEN - 1);
  int head = i >> 18;  // CH*LEN = 2^18
  float x = V[i] * colrcp[head * LEN + kcol];
  __bf16 h = (__bf16)x;
  Vh[i] = h;
  Vl[i] = (__bf16)(x - (float)h);
}

// ---------------------------------------------------------------------------
// 4) O = V' * exp(S - m)^T, recomputing S tiles. grid (LEN/128, NHEAD),
//    block 256. Each wave: one 16-wide q tile, 8 c-tile accumulators,
//    k-loop in steps of 32; P transposed via per-wave padded LDS.
//    NOTE: pbuf is wave-private, and LDS ops from one wave execute in order
//    in the DS pipe (ISA §7.1/§7.3), so no block barrier is needed in the
//    k-loop — waves run fully decoupled; only a compiler fence keeps the
//    store->load program order at the IR level.
// ---------------------------------------------------------------------------
__global__ __launch_bounds__(256) void attn_out(
    const __bf16* __restrict__ Qth, const __bf16* __restrict__ Qtl,
    const __bf16* __restrict__ Kth, const __bf16* __restrict__ Ktl,
    const __bf16* __restrict__ Vh, const __bf16* __restrict__ Vl,
    const float* __restrict__ colmax, float* __restrict__ O) {
  __shared__ float pbuf[8][16][36];  // [wave][q_local][k_local + pad]

  const int wave = threadIdx.x >> 5;
  const int lane = threadIdx.x & 31;
  const int lrow = lane & 15;
  const int lhi  = lane >> 4;
  const int head = blockIdx.y;
  const int q0   = blockIdx.x * 128 + wave * 16;
  const size_t hoff = (size_t)head * CH * LEN;
  const int hstat = head * LEN;

  // Hoisted Q^T A-fragments (16q x 128c)
  v16bf aqh[4], aql[4];
  const size_t qrow = hoff + (size_t)(q0 + lrow) * CH + lhi * 8;
#pragma unroll
  for (int cc = 0; cc < 4; ++cc) {
    aqh[cc] = loadA16(&Qth[qrow + cc * 32]);
    aql[cc] = loadA16(&Qtl[qrow + cc * 32]);
  }

  v8f o[8];
#pragma unroll
  for (int ct = 0; ct < 8; ++ct) {
    v8f z = ZERO8;
    o[ct] = z;
  }

  for (int k0 = 0; k0 < LEN; k0 += 32) {
    // prefetch next k-step's K rows (global_prefetch_b8; speculative, safe)
    __builtin_prefetch(&Kth[hoff + (size_t)(k0 + 32 + lrow) * CH], 0, 0);
    __builtin_prefetch(&Ktl[hoff + (size_t)(k0 + 32 + lrow) * CH], 0, 0);

    // --- recompute S tile [16q x 32k], exp(S - m[k]) -> LDS ---
#pragma unroll
    for (int t = 0; t < 2; ++t) {
      const int kc = k0 + t * 16;
      const size_t krow = hoff + (size_t)(kc + lrow) * CH;
      v8f s = ZERO8;
#pragma unroll
      for (int cc = 0; cc < 4; ++cc) {
        v16bf kbh = loadA16(&Kth[krow + cc * 32 + lhi * 8]);
        v16bf kbl = loadA16(&Ktl[krow + cc * 32 + lhi * 8]);
        s = wmma_split(aqh[cc], aql[cc], kbh, kbl, s);
      }
      float mc = colmax[hstat + kc + lrow];
#pragma unroll
      for (int r = 0; r < 8; ++r)
        pbuf[wave][lhi * 8 + r][t * 16 + lrow] = __expf(s[r] - mc);
    }
    // wave-private LDS + in-order DS pipe: compiler fence only, no barrier
    asm volatile("" ::: "memory");

    // B fragment = P^T (32k x 16q): lane -> q col lrow, elems kk = lhi*8+(e&7)+(e>>3)*16
    v16bf pbh, pbl;
#pragma unroll
    for (int e = 0; e < 16; ++e) {
      float x = pbuf[wave][lrow][lhi * 8 + (e & 7) + (e >> 3) * 16];
      __bf16 h = (__bf16)x;
      pbh[e] = h;
      pbl[e] = (__bf16)(x - (float)h);
    }

    // 8 c-tiles: A = V' (16c x 32k), element axis k contiguous in memory
#pragma unroll
    for (int ct = 0; ct < 8; ++ct) {
      const size_t vrow = hoff + (size_t)(ct * 16 + lrow) * LEN + k0 + lhi * 8;
      o[ct] = wmma_split(loadA16(&Vh[vrow]), loadA16(&Vl[vrow]),
                         pbh, pbl, o[ct]);
    }
  }

  // D tile: lane col q = q0+lrow, rows c = ct*16 + lhi*8 + r
#pragma unroll
  for (int ct = 0; ct < 8; ++ct) {
#pragma unroll
    for (int r = 0; r < 8; ++r) {
      int c = ct * 16 + lhi * 8 + r;
      O[hoff + (size_t)c * LEN + q0 + lrow] = o[ct][r];
    }
  }
}

extern "C" void kernel_launch(void* const* d_in, const int* in_sizes, int n_in,
                              void* d_out, int out_size, void* d_ws,
                              size_t ws_size, hipStream_t stream) {
  (void)in_sizes; (void)n_in; (void)out_size; (void)ws_size;
  const float* q = (const float*)d_in[0];
  const float* k = (const float*)d_in[1];
  const float* v = (const float*)d_in[2];
  float* out = (float*)d_out;

  const size_t NE = (size_t)NHEAD * CH * LEN;  // 8.4M elements
  char* w = (char*)d_ws;                        // needs ~101 MB
  __bf16* Qth = (__bf16*)w; w += NE * 2;
  __bf16* Qtl = (__bf16*)w; w += NE * 2;
  __bf16* Kth = (__bf16*)w; w += NE * 2;
  __bf16* Ktl = (__bf16*)w; w += NE * 2;
  __bf16* Vh  = (__bf16*)w; w += NE * 2;
  __bf16* Vl  = (__bf16*)w; w += NE * 2;
  float* colmax = (float*)w; w += (size_t)NHEAD * LEN * 4;
  float* colrcp = (float*)w;

  transpose_split<<<dim3(LEN / 64, NHEAD, 2), dim3(256), 0, stream>>>(
      q, k, Qth, Qtl, Kth, Ktl);
  qk_colstats<<<dim3(LEN / 128, NHEAD), dim3(256), 0, stream>>>(
      Qth, Qtl, Kth, Ktl, colmax, colrcp);
  v_scale_split<<<dim3((unsigned)(NE / 256)), dim3(256), 0, stream>>>(
      v, colrcp, Vh, Vl);
  attn_out<<<dim3(LEN / 128, NHEAD), dim3(256), 0, stream>>>(
      Qth, Qtl, Kth, Ktl, Vh, Vl, colmax, out);
}
